// SpatialAttention_70222715289848
// MI455X (gfx1250) — matrix operand order repaired and checked by reading, not compiled, not run on hardware
//
#include <hip/hip_runtime.h>
#include <hip/hip_fp16.h>

typedef __attribute__((ext_vector_type(16))) _Float16 v16h;
typedef __attribute__((ext_vector_type(8)))  float    v8f;

#define D_DIM 1024
#define SEQ   2048
#define BATCH 2
#define HEADS 16
#define HD    64
#define M_ROWS (BATCH * SEQ)   // 4096

union FragA { v16h h; unsigned int u[8]; };   // A-matrix: 8x 32b (K-pairs)
union FragB { v16h h; uint4 q[2]; };          // B-matrix: 2x 128b contiguous K runs

// ---------------------------------------------------------------------------
// CDNA5 async global -> LDS copy (ASYNCcnt-tracked DMA, no VGPR data path).
// LDS operand is the low 32 bits of the generic pointer (flat->LDS truncation).
// ---------------------------------------------------------------------------
__device__ __forceinline__ void g2l_async_b128(void* lds, const void* gaddr) {
    unsigned l = (unsigned)(unsigned long long)lds;
    asm volatile("global_load_async_to_lds_b128 %0, %1, off"
                 :: "v"(l), "v"(gaddr) : "memory");
}
__device__ __forceinline__ void wait_async0() {
    asm volatile("s_wait_asynccnt 0" ::: "memory");
}

// ---------------------------------------------------------------------------
// fp32 -> fp16 conversion (x)
// ---------------------------------------------------------------------------
__global__ __launch_bounds__(256)
void sa_cvt_f32_to_f16(const float* __restrict__ src, _Float16* __restrict__ dst, int n) {
    int i = (blockIdx.x * 256 + threadIdx.x) * 4;
    if (i + 3 < n) {
        float4 f = *(const float4*)(src + i);
        dst[i + 0] = (_Float16)f.x;
        dst[i + 1] = (_Float16)f.y;
        dst[i + 2] = (_Float16)f.z;
        dst[i + 3] = (_Float16)f.w;
    }
}

// ---------------------------------------------------------------------------
// fp32 W[k][n] -> fp16 WT[n][k] tiled transpose (32x32 tiles via LDS)
// ---------------------------------------------------------------------------
__global__ __launch_bounds__(256)
void sa_cvt_transpose(const float* __restrict__ Wq, const float* __restrict__ Wk,
                      const float* __restrict__ Wv,
                      _Float16* __restrict__ WTQ, _Float16* __restrict__ WTK,
                      _Float16* __restrict__ WTV) {
    const float* src; _Float16* dst;
    if (blockIdx.z == 0)      { src = Wq; dst = WTQ; }
    else if (blockIdx.z == 1) { src = Wk; dst = WTK; }
    else                      { src = Wv; dst = WTV; }

    __shared__ _Float16 t[32][33];
    const int n0 = blockIdx.x * 32, k0 = blockIdx.y * 32;
    const int tx = threadIdx.x & 31, ty = threadIdx.x >> 5;   // 8 rows per pass
    #pragma unroll
    for (int i = 0; i < 32; i += 8)
        t[ty + i][tx] = (_Float16)src[(size_t)(k0 + ty + i) * D_DIM + n0 + tx];
    __syncthreads();
    #pragma unroll
    for (int i = 0; i < 32; i += 8)
        dst[(size_t)(n0 + ty + i) * D_DIM + k0 + tx] = t[tx][ty + i];
}

// ---------------------------------------------------------------------------
// QKV projection GEMM: [4096 x 1024] x [1024 x 1024] + bias.
// B comes from pre-transposed WT[n][k] so both tiles stage as b128 async DMA.
// K-step 64, double-buffered LDS, 4 waves, wave tile 32x32.
// z=0 -> Q[B,H,S,64], z=1 -> K[B,H,S,64], z=2 -> V^T[B,H,64,S].
// ---------------------------------------------------------------------------
__global__ __launch_bounds__(128)
void sa_qkv_gemm(const _Float16* __restrict__ XH,
                 const _Float16* __restrict__ WTQ, const _Float16* __restrict__ WTK,
                 const _Float16* __restrict__ WTV,
                 const float* __restrict__ bq, const float* __restrict__ bk,
                 const float* __restrict__ bv,
                 _Float16* __restrict__ QH, _Float16* __restrict__ KH,
                 _Float16* __restrict__ VT) {
    const _Float16* WT; const float* bias;
    if (blockIdx.z == 0)      { WT = WTQ; bias = bq; }
    else if (blockIdx.z == 1) { WT = WTK; bias = bk; }
    else                      { WT = WTV; bias = bv; }

    __shared__ alignas(16) _Float16 Alds[2][64 * 72];   // [row][k], pitch 72
    __shared__ alignas(16) _Float16 Blds[2][64 * 72];   // [n][k],   pitch 72

    const int tid  = threadIdx.x;
    const int wave = tid >> 5, lane = tid & 31;
    const int hl   = lane & 15, hi = lane >> 4;
    const int wr   = wave >> 1, wc = wave & 1;
    const int row0 = blockIdx.y * 64;
    const int col0 = blockIdx.x * 64;

    // Preamble: async-issue tile 0 (64x64 halfs each for A and B)
    #pragma unroll
    for (int c = 0; c < 4; ++c) {
        int idx = tid + 128 * c;               // 0..511
        int r = idx >> 3, ck = (idx & 7) * 8;
        g2l_async_b128(&Alds[0][r * 72 + ck], XH + (size_t)(row0 + r) * D_DIM + ck);
        g2l_async_b128(&Blds[0][r * 72 + ck], WT + (size_t)(col0 + r) * D_DIM + ck);
    }

    v8f acc[2][2] = {};

    for (int i = 0; i < D_DIM / 64; ++i) {
        const int cur = i & 1, nxt = cur ^ 1;
        wait_async0();          // our wave's tile-i DMA done
        __syncthreads();        // everyone's tile-i written; tile i-1 consumers done
        if (i + 1 < D_DIM / 64) {
            const int k0n = (i + 1) * 64;
            #pragma unroll
            for (int c = 0; c < 4; ++c) {
                int idx = tid + 128 * c;
                int r = idx >> 3, ck = (idx & 7) * 8;
                g2l_async_b128(&Alds[nxt][r * 72 + ck],
                               XH + (size_t)(row0 + r) * D_DIM + k0n + ck);
                g2l_async_b128(&Blds[nxt][r * 72 + ck],
                               WT + (size_t)(col0 + r) * D_DIM + k0n + ck);
            }
        }
        // Compute on tile i (two 32-K steps)
        #pragma unroll
        for (int st = 0; st < 2; ++st) {
            FragA a[2]; FragB b[2];
            #pragma unroll
            for (int rg = 0; rg < 2; ++rg) {
                int m = wr * 32 + rg * 16 + hl;
                #pragma unroll
                for (int v = 0; v < 8; ++v) {
                    int kk = 32 * st + 2 * v + 16 * hi;
                    a[rg].u[v] = *(const unsigned int*)&Alds[cur][m * 72 + kk];
                }
            }
            #pragma unroll
            for (int cg = 0; cg < 2; ++cg) {
                int n = wc * 32 + cg * 16 + hl;
                const uint4* p = (const uint4*)&Blds[cur][n * 72 + 32 * st + 16 * hi];
                b[cg].q[0] = p[0]; b[cg].q[1] = p[1];
            }
            #pragma unroll
            for (int rg = 0; rg < 2; ++rg)
                #pragma unroll
                for (int cg = 0; cg < 2; ++cg)
                    acc[rg][cg] = __builtin_amdgcn_wmma_f32_16x16x32_f16(
                        false, a[rg].h, false, b[cg].h, (short)0, acc[rg][cg], false, false);
        }
    }

    // Epilogue: bias add (fp32) -> f16, scatter to head-major (V transposed)
    #pragma unroll
    for (int rg = 0; rg < 2; ++rg)
        #pragma unroll
        for (int cg = 0; cg < 2; ++cg)
            #pragma unroll
            for (int r = 0; r < 8; ++r) {
                int grow = row0 + wr * 32 + rg * 16 + r + 8 * hi;
                int gcol = col0 + wc * 32 + cg * 16 + hl;
                float v = acc[rg][cg][r] + bias[gcol];
                int bb = grow >> 11, s = grow & 2047;
                int h  = gcol >> 6,  d = gcol & 63;
                _Float16 hv = (_Float16)v;
                if (blockIdx.z == 0)
                    QH[(((size_t)(bb * HEADS + h)) * SEQ + s) * HD + d] = hv;
                else if (blockIdx.z == 1)
                    KH[(((size_t)(bb * HEADS + h)) * SEQ + s) * HD + d] = hv;
                else
                    VT[(((size_t)(bb * HEADS + h)) * HD + d) * SEQ + s] = hv;
            }
}

// ---------------------------------------------------------------------------
// Flash attention: grid (S/64, B*H), 128 threads, wave w owns 16 query rows.
// 32-key blocks, double-buffered async K/V staging, online softmax,
// no 1/sqrt(hd) scaling (matches reference).
// ---------------------------------------------------------------------------
__global__ __launch_bounds__(128)
void sa_flash_attn(const _Float16* __restrict__ QH, const _Float16* __restrict__ KH,
                   const _Float16* __restrict__ VT, float* __restrict__ out) {
    const int bh   = blockIdx.y;
    const int qblk = blockIdx.x;
    const _Float16* Q  = QH + (size_t)bh * SEQ * HD;
    const _Float16* K  = KH + (size_t)bh * SEQ * HD;
    const _Float16* Vt = VT + (size_t)bh * SEQ * HD;     // [HD][SEQ]
    const int b = bh >> 4, h = bh & 15;

    __shared__ alignas(16) _Float16 Qlds[64 * 72];       // [q][hd]   pitch 72
    __shared__ alignas(16) _Float16 Klds[2][32 * 72];    // [key][hd] pitch 72
    __shared__ alignas(16) _Float16 Vlds[2][64 * 40];    // [hd][key] pitch 40
    __shared__ alignas(16) _Float16 Plds[64 * 40];       // [q][key]  (wave-private rows)

    const int tid = threadIdx.x, wave = tid >> 5, lane = tid & 31;
    const int hl = lane & 15, hi = lane >> 4;

    // Preamble: async-stage Q tile and KV tile 0
    #pragma unroll
    for (int c = 0; c < 4; ++c) {
        int idx = tid + 128 * c;
        int r = idx >> 3, ck = (idx & 7) * 8;
        g2l_async_b128(&Qlds[r * 72 + ck], Q + (size_t)(qblk * 64 + r) * HD + ck);
    }
    #pragma unroll
    for (int c = 0; c < 2; ++c) {
        int idx = tid + 128 * c;                // 0..255
        int r = idx >> 3, ck = (idx & 7) * 8;
        g2l_async_b128(&Klds[0][r * 72 + ck], K + (size_t)r * HD + ck);
        int d = idx >> 2, cv = (idx & 3) * 8;
        g2l_async_b128(&Vlds[0][d * 40 + cv], Vt + (size_t)d * SEQ + cv);
    }
    wait_async0();
    __syncthreads();

    // Q A-frags (kept in registers for whole KV loop)
    FragA qf[2];
    #pragma unroll
    for (int st = 0; st < 2; ++st)
        #pragma unroll
        for (int v = 0; v < 8; ++v) {
            int kk = 32 * st + 2 * v + 16 * hi;
            qf[st].u[v] = *(const unsigned int*)&Qlds[(wave * 16 + hl) * 72 + kk];
        }

    v8f o0 = {}, o1 = {}, o2 = {}, o3 = {};
    float rmax[8], rsum[8];
    #pragma unroll
    for (int r = 0; r < 8; ++r) { rmax[r] = -1e30f; rsum[r] = 0.0f; }

    for (int i = 0; i < SEQ / 32; ++i) {
        const int cur = i & 1, nxt = cur ^ 1;
        wait_async0();          // our tile-i DMA done (no-op at i=0)
        __syncthreads();        // everyone's tile-i in LDS; tile i-1 consumers done
        if (i + 1 < SEQ / 32) {
            const int kbn = (i + 1) * 32;
            #pragma unroll
            for (int c = 0; c < 2; ++c) {
                int idx = tid + 128 * c;
                int r = idx >> 3, ck = (idx & 7) * 8;
                g2l_async_b128(&Klds[nxt][r * 72 + ck], K + (size_t)(kbn + r) * HD + ck);
                int d = idx >> 2, cv = (idx & 3) * 8;
                g2l_async_b128(&Vlds[nxt][d * 40 + cv], Vt + (size_t)d * SEQ + kbn + cv);
            }
            if (i + 2 < SEQ / 32)   // warm GL2 two tiles ahead
                __builtin_prefetch(K + (size_t)((i + 2) * 32) * HD + tid * 16, 0, 0);
        }

        // scores S = Q @ K^T : two 16x16 key groups, 2 WMMAs each over hd=64
        v8f s0 = {}, s1 = {};
        #pragma unroll
        for (int g = 0; g < 2; ++g) {
            v8f sc = {};
            #pragma unroll
            for (int st = 0; st < 2; ++st) {
                FragB kf;
                const uint4* p =
                    (const uint4*)&Klds[cur][(g * 16 + hl) * 72 + 32 * st + 16 * hi];
                kf.q[0] = p[0]; kf.q[1] = p[1];
                sc = __builtin_amdgcn_wmma_f32_16x16x32_f16(
                    false, qf[st].h, false, kf.h, (short)0, sc, false, false);
            }
            if (g == 0) s0 = sc; else s1 = sc;
        }

        // Online softmax; row m = wave*16 + r + 8*hi spans 16 lanes
        float pex0[8], pex1[8];
        #pragma unroll
        for (int r = 0; r < 8; ++r) {
            float m = fmaxf(s0[r], s1[r]);
            m = fmaxf(m, __shfl_xor(m, 1));
            m = fmaxf(m, __shfl_xor(m, 2));
            m = fmaxf(m, __shfl_xor(m, 4));
            m = fmaxf(m, __shfl_xor(m, 8));
            float newmax = fmaxf(rmax[r], m);
            float corr = __expf(rmax[r] - newmax);
            float e0 = __expf(s0[r] - newmax);
            float e1 = __expf(s1[r] - newmax);
            float ls = e0 + e1;
            ls += __shfl_xor(ls, 1);
            ls += __shfl_xor(ls, 2);
            ls += __shfl_xor(ls, 4);
            ls += __shfl_xor(ls, 8);
            rsum[r] = rsum[r] * corr + ls;
            rmax[r] = newmax;
            pex0[r] = e0; pex1[r] = e1;
            o0[r] *= corr; o1[r] *= corr; o2[r] *= corr; o3[r] *= corr;
        }

        // P: C-layout -> A-layout via wave-private LDS rows
        #pragma unroll
        for (int r = 0; r < 8; ++r) {
            int m = wave * 16 + r + 8 * hi;
            Plds[m * 40 + hl]      = (_Float16)pex0[r];
            Plds[m * 40 + 16 + hl] = (_Float16)pex1[r];
        }
        FragA pf;
        #pragma unroll
        for (int v = 0; v < 8; ++v) {
            int kk = 2 * v + 16 * hi;
            pf.u[v] = *(const unsigned int*)&Plds[(wave * 16 + hl) * 40 + kk];
        }

        // O += P @ V  (K-dim = 32 keys, 4 hd-column frags)
        #pragma unroll
        for (int c = 0; c < 4; ++c) {
            FragB vf;
            const uint4* p = (const uint4*)&Vlds[cur][(c * 16 + hl) * 40 + 16 * hi];
            vf.q[0] = p[0]; vf.q[1] = p[1];
            v8f& oc = (c == 0) ? o0 : (c == 1) ? o1 : (c == 2) ? o2 : o3;
            oc = __builtin_amdgcn_wmma_f32_16x16x32_f16(
                false, pf.h, false, vf.h, (short)0, oc, false, false);
        }
    }

    // Normalize and store fp32 output [B, S, D], col = h*64 + hd
    const int sbase = qblk * 64 + wave * 16;
    #pragma unroll
    for (int r = 0; r < 8; ++r) {
        float inv = 1.0f / rsum[r];
        int srow = sbase + r + 8 * hi;
        size_t base = ((size_t)b * SEQ + srow) * D_DIM + h * 64;
        out[base +  0 + hl] = o0[r] * inv;
        out[base + 16 + hl] = o1[r] * inv;
        out[base + 32 + hl] = o2[r] * inv;
        out[base + 48 + hl] = o3[r] * inv;
    }
}

// ---------------------------------------------------------------------------
extern "C" void kernel_launch(void* const* d_in, const int* in_sizes, int n_in,
                              void* d_out, int out_size, void* d_ws, size_t ws_size,
                              hipStream_t stream) {
    const float* x  = (const float*)d_in[0];
    const float* Wq = (const float*)d_in[1];
    const float* bq = (const float*)d_in[2];
    const float* Wk = (const float*)d_in[3];
    const float* bk = (const float*)d_in[4];
    const float* Wv = (const float*)d_in[5];
    const float* bv = (const float*)d_in[6];
    float* out = (float*)d_out;

    char* ws = (char*)d_ws;
    const size_t XH_B = (size_t)M_ROWS * D_DIM * 2;   // 8 MB
    const size_t W_B  = (size_t)D_DIM * D_DIM * 2;    // 2 MB
    _Float16* XH  = (_Float16*)(ws);
    _Float16* WTQ = (_Float16*)(ws + XH_B);
    _Float16* WTK = (_Float16*)(ws + XH_B + W_B);
    _Float16* WTV = (_Float16*)(ws + XH_B + 2 * W_B);
    _Float16* QH  = (_Float16*)(ws + XH_B + 3 * W_B);
    _Float16* KH  = (_Float16*)(ws + 2 * XH_B + 3 * W_B);
    _Float16* VT  = (_Float16*)(ws + 3 * XH_B + 3 * W_B);

    const int nx = M_ROWS * D_DIM;       // 4194304
    sa_cvt_f32_to_f16<<<nx / 1024, 256, 0, stream>>>(x, XH, nx);

    dim3 tgrid(D_DIM / 32, D_DIM / 32, 3);            // (32, 32, 3)
    sa_cvt_transpose<<<tgrid, 256, 0, stream>>>(Wq, Wk, Wv, WTQ, WTK, WTV);

    dim3 ggrid(D_DIM / 64, M_ROWS / 64, 3);           // (16, 64, 3)
    sa_qkv_gemm<<<ggrid, 128, 0, stream>>>(XH, WTQ, WTK, WTV, bq, bk, bv, QH, KH, VT);

    dim3 agrid(SEQ / 64, BATCH * HEADS);              // (32, 32)
    sa_flash_attn<<<agrid, 128, 0, stream>>>(QH, KH, VT, out);
}
